// FixedMaskAttention_15298673508442
// MI455X (gfx1250) — matrix operand order repaired and checked by reading, not compiled, not run on hardware
//
#include <hip/hip_runtime.h>
#include <hip/hip_bf16.h>
#include <math.h>

// ---------------------------------------------------------------------------
// Problem constants (match reference)
// ---------------------------------------------------------------------------
#define SEQ    2048
#define BATCH  4
#define HID    2048
#define NH     16
#define NKV    4
#define HD     128
#define NQKV   3072   // 2048 (Q) + 512 (K) + 512 (V) fused columns
#define KOFF   2048   // column offset of K in qkv
#define VOFF   2560   // column offset of V in qkv

typedef __attribute__((ext_vector_type(16))) __bf16 v16bf;
typedef __attribute__((ext_vector_type(8)))  float  v8f;

union Frag {
    v16bf v;
    uint4 q[2];
};

__device__ __forceinline__ v8f wmma_bf16(const Frag& a, const Frag& b, v8f c) {
    return __builtin_amdgcn_wmma_f32_16x16x32_bf16(
        /*neg_a=*/false, a.v, /*neg_b=*/false, b.v,
        /*c_mod=*/(short)0, c, /*reuse_a=*/false, /*reuse_b=*/false);
}

// CDNA5 async global->LDS copy (ASYNCcnt-tracked).  Low 32 bits of a generic
// shared pointer are the DS-style LDS byte address.
__device__ __forceinline__ void async_ld_b128(void* lds, const void* gaddr) {
    unsigned       ldsa = (unsigned)(uintptr_t)lds;
    unsigned long  ga   = (unsigned long)(uintptr_t)gaddr;
    asm volatile("global_load_async_to_lds_b128 %0, %1, off"
                 :: "v"(ldsa), "v"(ga) : "memory");
}
__device__ __forceinline__ void wait_async0() {
    asm volatile("s_wait_asynccnt 0x0" ::: "memory");
}

// ---------------------------------------------------------------------------
// 1) f32 -> bf16 bulk convert (8 elems / thread, 128-bit stores)
// ---------------------------------------------------------------------------
__global__ void k_f32_to_bf16(const float* __restrict__ in,
                              __bf16* __restrict__ out, int n8) {
    int i = blockIdx.x * blockDim.x + threadIdx.x;
    if (i >= n8) return;
    const float4* p = reinterpret_cast<const float4*>(in) + (size_t)i * 2;
    float4 a = p[0], b = p[1];
    union { __bf16 h[8]; uint4 u; } r;
    r.h[0] = (__bf16)a.x; r.h[1] = (__bf16)a.y;
    r.h[2] = (__bf16)a.z; r.h[3] = (__bf16)a.w;
    r.h[4] = (__bf16)b.x; r.h[5] = (__bf16)b.y;
    r.h[6] = (__bf16)b.z; r.h[7] = (__bf16)b.w;
    reinterpret_cast<uint4*>(out)[i] = r.u;
}

// ---------------------------------------------------------------------------
// 2) transpose-convert: in[K][N] f32  ->  out[N][K] bf16  (LDS 32x32 tile)
// ---------------------------------------------------------------------------
__global__ void k_transpose_bf16(const float* __restrict__ in,
                                 __bf16* __restrict__ out, int K, int N) {
    __shared__ float tile[32][33];
    const int k0 = blockIdx.y * 32, n0 = blockIdx.x * 32;
    const int tx = threadIdx.x, ty = threadIdx.y;
#pragma unroll
    for (int i = 0; i < 4; ++i)
        tile[ty + i * 8][tx] = in[(size_t)(k0 + ty + i * 8) * N + n0 + tx];
    __syncthreads();
#pragma unroll
    for (int i = 0; i < 4; ++i)
        out[(size_t)(n0 + ty + i * 8) * K + k0 + tx] =
            (__bf16)tile[tx][ty + i * 8];
}

// ---------------------------------------------------------------------------
// 3) bf16 WMMA GEMM:  C[M][N] = A[M][K] * B^T[N][K].
//    Block tile 128x256, 8 waves (2 M x 4 N), per-wave 64x64 (4x4 WMMA).
//    Register double-buffer: next K-slice global loads issued before the
//    WMMA burst so HBM/L2 latency hides under compute.
// ---------------------------------------------------------------------------
template <typename OutT>
__global__ __launch_bounds__(256)
void k_gemm_bf16(const __bf16* __restrict__ A, const __bf16* __restrict__ BT,
                 OutT* __restrict__ C, int M, int N, int K) {
    __shared__ __align__(16) __bf16 As[128 * 40];   // [m][k]  ld=40 (pad)
    __shared__ __align__(16) __bf16 Bs[256 * 40];   // [n][k]  ld=40

    const int tid  = threadIdx.x;
    const int lane = tid & 31;
    const int wv   = tid >> 5;
    const int wm   = wv >> 2;        // 0..1  (M dir)
    const int wn   = wv & 3;         // 0..3  (N dir)
    const int hf   = lane >> 4;      // lane half
    const int l15  = lane & 15;
    const int m0   = blockIdx.y * 128;
    const int n0   = blockIdx.x * 256;

    v8f acc[4][4];
    const v8f vzero = {0.f, 0.f, 0.f, 0.f, 0.f, 0.f, 0.f, 0.f};
#pragma unroll
    for (int mt = 0; mt < 4; ++mt)
#pragma unroll
        for (int nt = 0; nt < 4; ++nt) acc[mt][nt] = vzero;

    uint4 ra[2], rb[4];
    auto load_tiles = [&](int kb) {
#pragma unroll
        for (int i = 0; i < 2; ++i) {        // A: 128x32, 2 chunks/thread
            int c = tid * 2 + i, r = c >> 2, cc = c & 3;
            ra[i] = *reinterpret_cast<const uint4*>(
                A + (size_t)(m0 + r) * K + kb + cc * 8);
        }
#pragma unroll
        for (int i = 0; i < 4; ++i)          // B: 256x32, row = tid
            rb[i] = *reinterpret_cast<const uint4*>(
                BT + (size_t)(n0 + tid) * K + kb + i * 8);
    };

    load_tiles(0);
    for (int kb = 0; kb < K; kb += 32) {
        __syncthreads();                     // readers of previous tile done
#pragma unroll
        for (int i = 0; i < 2; ++i) {
            int c = tid * 2 + i, r = c >> 2, cc = c & 3;
            *reinterpret_cast<uint4*>(&As[r * 40 + cc * 8]) = ra[i];
        }
#pragma unroll
        for (int i = 0; i < 4; ++i)
            *reinterpret_cast<uint4*>(&Bs[tid * 40 + i * 8]) = rb[i];
        __syncthreads();
        if (kb + 32 < K) load_tiles(kb + 32);   // in flight during WMMAs

        // A frag (16-bit 16x32): lane<16 -> K[0..15]; lane>=16 -> K[16..31]
        Frag a[4];
#pragma unroll
        for (int mt = 0; mt < 4; ++mt) {
            int r = wm * 64 + mt * 16 + l15;
            a[mt].q[0] = *reinterpret_cast<const uint4*>(&As[r * 40 + hf * 16]);
            a[mt].q[1] = *reinterpret_cast<const uint4*>(&As[r * 40 + hf * 16 + 8]);
        }
        // B frag (16-bit 32x16): lane<16 -> K[0..15]; lane>=16 -> K[16..31]
#pragma unroll
        for (int nt = 0; nt < 4; ++nt) {
            Frag b;
            int r = wn * 64 + nt * 16 + l15;
            b.q[0] = *reinterpret_cast<const uint4*>(&Bs[r * 40 + hf * 16]);
            b.q[1] = *reinterpret_cast<const uint4*>(&Bs[r * 40 + hf * 16 + 8]);
#pragma unroll
            for (int mt = 0; mt < 4; ++mt)
                acc[mt][nt] = wmma_bf16(a[mt], b, acc[mt][nt]);
        }
    }

    // C layout: vgpr v -> row v (lanes 0-15) / v+8 (lanes 16-31), col = l15
#pragma unroll
    for (int mt = 0; mt < 4; ++mt)
#pragma unroll
        for (int nt = 0; nt < 4; ++nt)
#pragma unroll
            for (int v = 0; v < 8; ++v) {
                int row = m0 + wm * 64 + mt * 16 + v + hf * 8;
                int col = n0 + wn * 64 + nt * 16 + l15;
                C[(size_t)row * N + col] = (OutT)acc[mt][nt][v];
            }
}

// ---------------------------------------------------------------------------
// 4) RoPE (in place on a head-slice of qkv).  One thread per (row, head, j<64)
// ---------------------------------------------------------------------------
__global__ void k_rope(__bf16* __restrict__ x, const int* __restrict__ pos_ids,
                       int nheads, int col_off) {
    int t = blockIdx.x * blockDim.x + threadIdx.x;
    int per_row = nheads * 64;
    int row = t / per_row;
    if (row >= BATCH * SEQ) return;
    int rem  = t - row * per_row;
    int head = rem >> 6;
    int j    = rem & 63;
    int b    = row >> 11;                 // /SEQ
    int s    = row & (SEQ - 1);
    float pos = (float)pos_ids[(b & 1) * SEQ + s];
    float inv = __powf(10000.0f, -(float)(2 * j) * (1.0f / 128.0f));
    float ang = pos * inv;
    float c = __cosf(ang), sn = __sinf(ang);
    size_t base = (size_t)row * NQKV + col_off + head * HD + j;
    float x1 = (float)x[base], x2 = (float)x[base + 64];
    x[base]      = (__bf16)(x1 * c - x2 * sn);
    x[base + 64] = (__bf16)(x2 * c + x1 * sn);
}

// ---------------------------------------------------------------------------
// 5) Flash attention.  Grid (SEQ/64, NH, BATCH), 128 threads (4 waves).
//    Q/K tiles staged with GLOBAL_LOAD_ASYNC_TO_LDS_B128 (ASYNCcnt);
//    V staged through VGPRs (needs in-flight transpose).
// ---------------------------------------------------------------------------
__global__ __launch_bounds__(128)
void k_attn(const __bf16* __restrict__ qkv, __bf16* __restrict__ attn) {
    __shared__ __align__(16) __bf16 Qs[64 * 136];    // [qrow][d]    ld=136
    __shared__ __align__(16) __bf16 Ks[32 * 136];    // [kpos][d]    (== B^T)
    __shared__ __align__(16) __bf16 Vt[128 * 40];    // [d][kpos]    (== B^T)
    __shared__ __align__(16) __bf16 Ps[4 * 16 * 40]; // per-wave P, [row][kpos]

    const int tid  = threadIdx.x;
    const int lane = tid & 31;
    const int w    = tid >> 5;
    const int hf   = lane >> 4;
    const int l15  = lane & 15;
    const int head = blockIdx.y;
    const int bz   = blockIdx.z;
    const int hkv  = head >> 2;           // GQA: 4 Q heads per KV head
    const bool causal = (bz < 2);
    const int qrow0 = blockIdx.x * 64;

    // stage Q block (64x128) asynchronously
#pragma unroll
    for (int i = 0; i < 8; ++i) {
        int c = tid + i * 128;            // 0..1023
        int r = c >> 4, cc = c & 15;
        async_ld_b128(&Qs[r * 136 + cc * 8],
                      qkv + (size_t)(bz * SEQ + qrow0 + r) * NQKV + head * HD + cc * 8);
    }

    const v8f vzero = {0.f, 0.f, 0.f, 0.f, 0.f, 0.f, 0.f, 0.f};
    v8f o[8];
#pragma unroll
    for (int t = 0; t < 8; ++t) o[t] = vzero;
    float m_i[8], l_i[8];
#pragma unroll
    for (int v = 0; v < 8; ++v) { m_i[v] = -3.0e38f; l_i[v] = 0.f; }

    const int nkb = causal ? (qrow0 + 64) / 32 : (SEQ / 32);
    for (int kbi = 0; kbi < nkb; ++kbi) {
        const int kp0 = kbi * 32;
        // stage K (async, direct) and V (VGPR path, transposed)
#pragma unroll
        for (int i = 0; i < 4; ++i) {
            int c = tid + i * 128;        // 0..511
            int r = c >> 4, cc = c & 15;
            async_ld_b128(&Ks[r * 136 + cc * 8],
                          qkv + (size_t)(bz * SEQ + kp0 + r) * NQKV + KOFF + hkv * HD + cc * 8);
            union { uint4 u; __bf16 h[8]; } vv;
            vv.u = *reinterpret_cast<const uint4*>(
                qkv + (size_t)(bz * SEQ + kp0 + r) * NQKV + VOFF + hkv * HD + cc * 8);
#pragma unroll
            for (int j = 0; j < 8; ++j) Vt[(cc * 8 + j) * 40 + r] = vv.h[j];
        }
        wait_async0();        // our async Q/K copies landed in LDS
        __syncthreads();      // everyone's staging visible

        // scores: Q(16x128) @ K^T(128x32) over 4 WMMA K-steps
        v8f s0 = vzero, s1 = vzero;
#pragma unroll
        for (int ds = 0; ds < 4; ++ds) {
            int ko = ds * 32;
            Frag a, b0, b1;
            int qr = w * 16 + l15;
            a.q[0]  = *reinterpret_cast<const uint4*>(&Qs[qr * 136 + ko + hf * 16]);
            a.q[1]  = *reinterpret_cast<const uint4*>(&Qs[qr * 136 + ko + hf * 16 + 8]);
            b0.q[0] = *reinterpret_cast<const uint4*>(&Ks[l15 * 136 + ko + hf * 16]);
            b0.q[1] = *reinterpret_cast<const uint4*>(&Ks[l15 * 136 + ko + hf * 16 + 8]);
            b1.q[0] = *reinterpret_cast<const uint4*>(&Ks[(16 + l15) * 136 + ko + hf * 16]);
            b1.q[1] = *reinterpret_cast<const uint4*>(&Ks[(16 + l15) * 136 + ko + hf * 16 + 8]);
            s0 = wmma_bf16(a, b0, s0);
            s1 = wmma_bf16(a, b1, s1);
        }

        // online softmax (rows live in 16-lane halves; xor masks 1..8 stay inside)
        float alpha[8];
#pragma unroll
        for (int v = 0; v < 8; ++v) {
            const float scale = 0.08838834764831845f;  // 1/sqrt(128)
            float f0 = s0[v] * scale, f1 = s1[v] * scale;
            int grow = qrow0 + w * 16 + v + hf * 8;
            int c0 = kp0 + l15, c1 = c0 + 16;
            if (causal) {
                if (c0 > grow) f0 = -3.0e38f;
                if (c1 > grow) f1 = -3.0e38f;
            }
            float rm = fmaxf(f0, f1);
#pragma unroll
            for (int mk = 1; mk < 16; mk <<= 1)
                rm = fmaxf(rm, __shfl_xor(rm, mk, 32));
            float newm = fmaxf(m_i[v], rm);
            float al = __expf(m_i[v] - newm);
            float p0 = __expf(f0 - newm);
            float p1 = __expf(f1 - newm);
            float rs = p0 + p1;
#pragma unroll
            for (int mk = 1; mk < 16; mk <<= 1)
                rs += __shfl_xor(rs, mk, 32);
            l_i[v] = l_i[v] * al + rs;
            m_i[v] = newm;
            alpha[v] = al;
            int pr = w * 640 + (v + hf * 8) * 40 + l15;
            Ps[pr]      = (__bf16)p0;
            Ps[pr + 16] = (__bf16)p1;
        }
#pragma unroll
        for (int t = 0; t < 8; ++t)
#pragma unroll
            for (int v = 0; v < 8; ++v) o[t][v] *= alpha[v];

        // out += P(16x32) @ V(32x128); same-wave DS store->load is in order
        Frag a2;
        a2.q[0] = *reinterpret_cast<const uint4*>(&Ps[w * 640 + l15 * 40 + hf * 16]);
        a2.q[1] = *reinterpret_cast<const uint4*>(&Ps[w * 640 + l15 * 40 + hf * 16 + 8]);
#pragma unroll
        for (int t = 0; t < 8; ++t) {
            Frag b2;
            int vr = t * 16 + l15;
            b2.q[0] = *reinterpret_cast<const uint4*>(&Vt[vr * 40 + hf * 16]);
            b2.q[1] = *reinterpret_cast<const uint4*>(&Vt[vr * 40 + hf * 16 + 8]);
            o[t] = wmma_bf16(a2, b2, o[t]);
        }
        __syncthreads();   // protect Ks/Vt before next block's stage
    }

    // normalize and write bf16 attention output [b*S + row][head*128 + d]
#pragma unroll
    for (int t = 0; t < 8; ++t)
#pragma unroll
        for (int v = 0; v < 8; ++v) {
            int grow = qrow0 + w * 16 + v + hf * 8;
            float val = o[t][v] / l_i[v];
            attn[(size_t)(bz * SEQ + grow) * HID + head * HD + t * 16 + l15] =
                (__bf16)val;
        }
}

// ---------------------------------------------------------------------------
// Launch
// ---------------------------------------------------------------------------
extern "C" void kernel_launch(void* const* d_in, const int* in_sizes, int n_in,
                              void* d_out, int out_size, void* d_ws, size_t ws_size,
                              hipStream_t stream) {
    const float* h   = (const float*)d_in[0];
    const float* Wq  = (const float*)d_in[1];
    const float* Wk  = (const float*)d_in[2];
    const float* Wv  = (const float*)d_in[3];
    const float* Wo  = (const float*)d_in[4];
    const int*   pos = (const int*)d_in[5];
    float* out = (float*)d_out;

    char* ws = (char*)d_ws;
    __bf16* hB   = (__bf16*)(ws);                 // 8192x2048       (33.6 MB)
    __bf16* WT   = (__bf16*)(ws + 33554432);      // 3072x2048 B^T   (12.6 MB)
    __bf16* WoT  = (__bf16*)(ws + 46137344);      // 2048x2048 B^T   ( 8.4 MB)
    __bf16* qkv  = (__bf16*)(ws + 54525952);      // 8192x3072       (50.3 MB)
    __bf16* attn = (__bf16*)(ws + 104857600);     // 8192x2048       (33.6 MB)

    // 1) h -> bf16
    int n8 = (BATCH * SEQ * HID) / 8;
    k_f32_to_bf16<<<n8 / 256, 256, 0, stream>>>(h, hB, n8);

    // 2) weight transpose-converts into fused qkv B^T and Wo B^T
    dim3 tb(32, 8);
    k_transpose_bf16<<<dim3(2048 / 32, 2048 / 32), tb, 0, stream>>>(Wq, WT, 2048, 2048);
    k_transpose_bf16<<<dim3(512 / 32, 2048 / 32), tb, 0, stream>>>(
        Wk, WT + (size_t)2048 * 2048, 2048, 512);
    k_transpose_bf16<<<dim3(512 / 32, 2048 / 32), tb, 0, stream>>>(
        Wv, WT + (size_t)2560 * 2048, 2048, 512);
    k_transpose_bf16<<<dim3(2048 / 32, 2048 / 32), tb, 0, stream>>>(Wo, WoT, 2048, 2048);

    // 3) fused QKV projection: (8192x2048) x (2048x3072) -> bf16
    k_gemm_bf16<__bf16><<<dim3(NQKV / 256, (BATCH * SEQ) / 128), 256, 0, stream>>>(
        hB, WT, qkv, BATCH * SEQ, NQKV, HID);

    // 4) RoPE on Q and K slices (in place)
    k_rope<<<(BATCH * SEQ * NH * 64) / 256, 256, 0, stream>>>(qkv, pos, NH, 0);
    k_rope<<<(BATCH * SEQ * NKV * 64) / 256, 256, 0, stream>>>(qkv, pos, NKV, KOFF);

    // 5) flash attention -> bf16 [8192 x 2048]
    k_attn<<<dim3(SEQ / 64, NH, BATCH), 128, 0, stream>>>(qkv, attn);

    // 6) output projection -> f32 d_out
    k_gemm_bf16<float><<<dim3(HID / 256, (BATCH * SEQ) / 128), 256, 0, stream>>>(
        attn, WoT, out, BATCH * SEQ, HID, HID);
}
